// CNN_pre_LSTM_26817775796508
// MI455X (gfx1250) — matrix operand order repaired and checked by reading, not compiled, not run on hardware
//
#include <hip/hip_runtime.h>

// CDNA5 (gfx1250) wave32 WMMA implementation of the small per-sample CNN.
// Each wave processes 16 samples; convs 2/3/4 and the final linear are
// v_wmma_f32_16x16x32_f16 GEMMs over im2col'd activations staged in LDS (f16).
//
// Key points:
//  - A-fragment slots whose K index has no weight are covered by zeros in the
//    *B* fragment, so all A loads are unconditional (no EXEC divergence) as
//    long as they stay inside the zero-padded LDS rows.
//  - maxpool pairs are computed as two WMMA accumulators in the SAME lane,
//    so pooling is a register max (no cross-lane traffic, no ds_bpermute).

typedef __attribute__((ext_vector_type(16))) _Float16 v16h;
typedef __attribute__((ext_vector_type(8)))  _Float16 v8h;
typedef __attribute__((ext_vector_type(8)))  float    v8f;

union AFrag { v16h v; v8h h[2]; };

#define WMMA_F32_F16(A, B, C) \
  __builtin_amdgcn_wmma_f32_16x16x32_f16(false, (A), false, (B), (short)0, (C), false, false)

// ---- A-fragment builders (ISA 7.12.2: 16-bit A 16x32, row M = lane&15,
//      element j -> K = 8*hi + (j&7) + 16*(j>>3)) --------------------------

// 8-channel input, K-order k = t*8 + ci. Upper half at hi==1 maps to
// K=24..31 where B is zero -> in-bounds garbage is harmless.
__device__ __forceinline__ v16h load_a8(const _Float16* __restrict__ base,
                                        int m, int pitch, int p, int hi) {
  AFrag f;
  f.h[0] = *reinterpret_cast<const v8h*>(base + (m * pitch + p + hi) * 8);
  f.h[1] = *reinterpret_cast<const v8h*>(base + (m * pitch + p + 2) * 8);
  return f.v;
}

// 16-channel input (layer 4), K-order k = t*16 + ci, split into two K-chunks.
__device__ __forceinline__ v16h load_a16_c0(const _Float16* __restrict__ base,
                                            int m, int p, int hi) {
  AFrag f;                       // kg 0..31  (t = 0, 1)
  f.h[0] = *reinterpret_cast<const v8h*>(base + (m * 14 + p    ) * 16 + 8 * hi);
  f.h[1] = *reinterpret_cast<const v8h*>(base + (m * 14 + p + 1) * 16 + 8 * hi);
  return f.v;
}
__device__ __forceinline__ v16h load_a16_c1(const _Float16* __restrict__ base,
                                            int m, int p, int hi) {
  AFrag f;                       // kg 32..47 (t = 2); kg>=48 -> B is zero
  f.h[0] = *reinterpret_cast<const v8h*>(base + (m * 14 + p + 2) * 16 + 8 * hi);
  v8h z = {};
  f.h[1] = z;
  return f.v;
}

// Flattened 96-feature input for the linear layer.
__device__ __forceinline__ v16h load_a_feat(const _Float16* __restrict__ feat,
                                            int m, int ch, int hi) {
  AFrag f;
  f.h[0] = *reinterpret_cast<const v8h*>(feat + m * 96 + ch * 32 + 8 * hi);
  f.h[1] = *reinterpret_cast<const v8h*>(feat + m * 96 + ch * 32 + 16 + 8 * hi);
  return f.v;
}

__global__ __launch_bounds__(128, 1)
void cnn_pre_lstm_kernel(const float* __restrict__ x,
                         const float* __restrict__ w11, const float* __restrict__ b11,
                         const float* __restrict__ w12, const float* __restrict__ b12,
                         const float* __restrict__ w21, const float* __restrict__ b21,
                         const float* __restrict__ w22, const float* __restrict__ b22,
                         const float* __restrict__ Wl,  const float* __restrict__ bl,
                         float* __restrict__ out)
{
  // Per-wave LDS slices (f16), two aliased regions:
  //  region A: h1 [16][26][8] (3328 h)  then  h3 [16][14][16] (3584 h)
  //  region B: h2p [16][14][8] (1792 h) then  feat [16][96]   (1536 h)
  constexpr int REGA = 3584;
  constexpr int REGB = 1792;
  constexpr int WAVE_SLAB = REGA + REGB;          // 5376 halves = 10752 B
  __shared__ __align__(16) _Float16 slab[4 * WAVE_SLAB];

  const int tid  = threadIdx.x;
  const int lane = tid & 31;
  const int wave = tid >> 5;
  const int lw   = lane & 15;      // N / M-row index inside fragments
  const int hi   = lane >> 4;      // lane half
  const int sampleBase = blockIdx.x * 64 + wave * 16;

  _Float16* regA = slab + wave * WAVE_SLAB;
  _Float16* regB = regA + REGA;

  // Each lane owns sample m = lw, position block pBase..pBase+11 for layer 1.
  const int pBase = 12 * hi;
  const float* xp = x + (sampleBase + lw) * 24;
  __builtin_prefetch(xp + pBase, 0, 0);   // global_prefetch_b8: overlap HBM with LDS zeroing

  // ---------------- zero LDS slab (provides all zero-padding) -------------
  {
    v8h z = {};
    for (int i = tid; i < (4 * WAVE_SLAB) / 8; i += 128)
      reinterpret_cast<v8h*>(slab)[i] = z;
  }
  __syncthreads();

  // ---------------- Layer 1: conv 1->8, K=3, VALU (K too thin for WMMA) ---
  {
    float W[8][3], B[8];
    for (int co = 0; co < 8; ++co) {
      W[co][0] = w11[co * 3 + 0];
      W[co][1] = w11[co * 3 + 1];
      W[co][2] = w11[co * 3 + 2];
      B[co]    = b11[co];
    }
    // Sliding window: 14 inputs -> 12 outputs per lane, loaded as 3x float4.
    float xv[14];
    xv[0]  = hi ? xp[11] : 0.f;
    const float4* xq = reinterpret_cast<const float4*>(xp + pBase);
    float4 q0 = xq[0], q1 = xq[1], q2 = xq[2];
    xv[1] = q0.x;  xv[2] = q0.y;  xv[3]  = q0.z;  xv[4]  = q0.w;
    xv[5] = q1.x;  xv[6] = q1.y;  xv[7]  = q1.z;  xv[8]  = q1.w;
    xv[9] = q2.x;  xv[10] = q2.y; xv[11] = q2.z;  xv[12] = q2.w;
    xv[13] = hi ? 0.f : xp[12];
    for (int p = 0; p < 12; ++p) {
      v8h o;
#pragma unroll
      for (int co = 0; co < 8; ++co) {
        float v = fmaf(W[co][0], xv[p],
                  fmaf(W[co][1], xv[p + 1],
                  fmaf(W[co][2], xv[p + 2], B[co])));
        o[co] = (_Float16)fmaxf(v, 0.f);
      }
      *reinterpret_cast<v8h*>(regA + (lw * 26 + pBase + p + 1) * 8) = o;  // h1
    }
  }
  __syncthreads();

  // ---------------- Layer 2: conv 8->8 (K=24) WMMA + maxpool2 ------------
  // Two WMMAs per pooled pair; both accumulators live in the same lane, so
  // the pool is a pure register max and the store predicate is hoisted.
  {
    v16h Bw;                      // B element j -> K = 8*hi + (j&7) + 16*(j>>3), N = lw
#pragma unroll
    for (int j = 0; j < 16; ++j) {
      int k = 8 * hi + (j & 7) + 16 * (j >> 3);        // k = t*8 + ci
      float v = 0.f;
      if (lw < 8 && k < 24) v = w12[(lw * 8 + (k & 7)) * 3 + (k >> 3)];
      Bw[j] = (_Float16)v;
    }
    float bias = (lw < 8) ? b12[lw] : 0.f;
    for (int pq = 0; pq < 12; ++pq) {
      v8f c0 = {}, c1 = {};
      c0 = WMMA_F32_F16(load_a8(regA, lw, 26, 2 * pq,     hi), Bw, c0);
      c1 = WMMA_F32_F16(load_a8(regA, lw, 26, 2 * pq + 1, hi), Bw, c1);
      if (lw < 8) {               // single EXEC toggle per pooled position
#pragma unroll
        for (int j = 0; j < 8; ++j) {
          float v = fmaxf(fmaxf(c0[j], c1[j]) + bias, 0.f);
          int m = j + 8 * hi;
          regB[(m * 14 + pq + 1) * 8 + lw] = (_Float16)v;   // h2p
        }
      }
    }
  }
  __syncthreads();

  // ---------------- Layer 3: conv 8->16 (K=24) WMMA ----------------------
  {
    // zero h3 pad rows (pp=0 and pp=13) before interior writes
    v8h z = {};
    for (int i = lane; i < 64; i += 32) {
      int m = i >> 2, r = i & 3;
      int pp = (r >> 1) ? 13 : 0;
      int co = (r & 1) * 8;
      *reinterpret_cast<v8h*>(regA + (m * 14 + pp) * 16 + co) = z;
    }
    v16h Bw;
#pragma unroll
    for (int j = 0; j < 16; ++j) {
      int k = 8 * hi + (j & 7) + 16 * (j >> 3);        // k = t*8 + ci
      float v = (k < 24) ? w21[(lw * 8 + (k & 7)) * 3 + (k >> 3)] : 0.f;
      Bw[j] = (_Float16)v;
    }
    float bias = b21[lw];
    for (int p = 0; p < 12; ++p) {
      v8f c = {};
      c = WMMA_F32_F16(load_a8(regB, lw, 14, p, hi), Bw, c);
#pragma unroll
      for (int j = 0; j < 8; ++j) {
        float v = fmaxf(c[j] + bias, 0.f);
        int m = j + 8 * hi;
        regA[(m * 14 + p + 1) * 16 + lw] = (_Float16)v;     // h3
      }
    }
  }
  __syncthreads();

  // ---------------- Layer 4: conv 16->16 (K=48, 2 chunks) + maxpool2 -----
  {
    v16h Bw0, Bw1;
#pragma unroll
    for (int j = 0; j < 16; ++j) {
      int k0 = 8 * hi + (j & 7) + 16 * (j >> 3);          // kg for chunk 0
      int k1 = 32 + 8 * hi + (j & 7) + 16 * (j >> 3);     // kg for chunk 1
      Bw0[j] = (_Float16)((k0 < 48) ? w22[(lw * 16 + (k0 & 15)) * 3 + (k0 >> 4)] : 0.f);
      Bw1[j] = (_Float16)((k1 < 48) ? w22[(lw * 16 + (k1 & 15)) * 3 + (k1 >> 4)] : 0.f);
    }
    float bias = b22[lw];
    for (int pq = 0; pq < 6; ++pq) {
      v8f c0 = {}, c1 = {};
      c0 = WMMA_F32_F16(load_a16_c0(regA, lw, 2 * pq,     hi), Bw0, c0);
      c0 = WMMA_F32_F16(load_a16_c1(regA, lw, 2 * pq,     hi), Bw1, c0);
      c1 = WMMA_F32_F16(load_a16_c0(regA, lw, 2 * pq + 1, hi), Bw0, c1);
      c1 = WMMA_F32_F16(load_a16_c1(regA, lw, 2 * pq + 1, hi), Bw1, c1);
#pragma unroll
      for (int j = 0; j < 8; ++j) {
        float v = fmaxf(fmaxf(c0[j], c1[j]) + bias, 0.f);
        int m = j + 8 * hi;
        regB[m * 96 + lw * 6 + pq] = (_Float16)v;   // torch flatten: feat = c*6 + p
      }
    }
  }
  __syncthreads();

  // ---------------- Linear 96 -> 24 (WMMA, 3 K-chunks x 2 N-tiles) -------
  {
    for (int nt = 0; nt < 2; ++nt) {
      int ng = nt * 16 + lw;
      v16h B0, B1, B2;
#pragma unroll
      for (int j = 0; j < 16; ++j) {
        int k = 8 * hi + (j & 7) + 16 * (j >> 3);
        B0[j] = (_Float16)((ng < 24) ? Wl[ng * 96 + k]      : 0.f);
        B1[j] = (_Float16)((ng < 24) ? Wl[ng * 96 + 32 + k] : 0.f);
        B2[j] = (_Float16)((ng < 24) ? Wl[ng * 96 + 64 + k] : 0.f);
      }
      v8f c = {};
      c = WMMA_F32_F16(load_a_feat(regB, lw, 0, hi), B0, c);
      c = WMMA_F32_F16(load_a_feat(regB, lw, 1, hi), B1, c);
      c = WMMA_F32_F16(load_a_feat(regB, lw, 2, hi), B2, c);
      if (ng < 24) {
        float bias = bl[ng];
#pragma unroll
        for (int j = 0; j < 8; ++j) {
          int m = j + 8 * hi;
          out[(sampleBase + m) * 24 + ng] = c[j] + bias;
        }
      }
    }
  }
}

extern "C" void kernel_launch(void* const* d_in, const int* in_sizes, int n_in,
                              void* d_out, int out_size, void* d_ws, size_t ws_size,
                              hipStream_t stream) {
  (void)in_sizes; (void)n_in; (void)d_ws; (void)ws_size; (void)out_size;
  const float* x   = (const float*)d_in[0];
  const float* w11 = (const float*)d_in[1];
  const float* b11 = (const float*)d_in[2];
  const float* w12 = (const float*)d_in[3];
  const float* b12 = (const float*)d_in[4];
  const float* w21 = (const float*)d_in[5];
  const float* b21 = (const float*)d_in[6];
  const float* w22 = (const float*)d_in[7];
  const float* b22 = (const float*)d_in[8];
  const float* Wl  = (const float*)d_in[9];
  const float* bl  = (const float*)d_in[10];
  float* out = (float*)d_out;

  // 512*256 = 131072 samples; 64 samples per block (4 waves x 16) -> 2048 blocks
  dim3 grid(2048), block(128);
  cnn_pre_lstm_kernel<<<grid, block, 0, stream>>>(x, w11, b11, w12, b12,
                                                  w21, b21, w22, b22, Wl, bl, out);
}
